// PeptideSelfAttention_57664230916412
// MI455X (gfx1250) — matrix operand order repaired and checked by reading, not compiled, not run on hardware
//
#include <hip/hip_runtime.h>
#include <hip/hip_bf16.h>

#define B_    16
#define N_    256
#define CS_   256
#define CH_   64
#define H_    8
#define KCLIP 32
#define CT_   1024
#define NB_   65
#define CATD  ((NB_ + CH_) * H_)   // 1032
#define QKN   (CH_ * H_)           // 512

typedef __attribute__((ext_vector_type(16))) _Float16 v16h;
typedef __attribute__((ext_vector_type(8)))  _Float16 h8;
typedef __attribute__((ext_vector_type(4)))  _Float16 hx4;
typedef __attribute__((ext_vector_type(8)))  float    v8f;
typedef __attribute__((ext_vector_type(4)))  float    fx4;

__device__ __forceinline__ v8f wmma16(v16h a, v16h b, v8f c) {
    return __builtin_amdgcn_wmma_f32_16x16x32_f16(false, a, false, b, (short)0, c,
                                                  false, false);
}
__device__ __forceinline__ v16h frag_cat(h8 lo, h8 hi) {
    return __builtin_shufflevector(lo, hi, 0, 1, 2, 3, 4, 5, 6, 7,
                                   8, 9, 10, 11, 12, 13, 14, 15);
}
__device__ __forceinline__ v8f v8f_zero() {
    v8f z;
#pragma unroll
    for (int t = 0; t < 8; ++t) z[t] = 0.f;
    return z;
}

// ---------------------------------------------------------------------------
// GEMM: C[M,Nc] = A[M,K] @ B[K,Nc] (+bias, optional ReLU).
// Block tile 128x128, 8 waves (4x2), each wave 32x64 = 8 wmma accumulators.
// M mult of 128, Nc mult of 128. Full K-steps use unchecked float4 loads;
// one checked tail step handles K % 32 (K=1032 for the Wo GEMM).
// permute_head: scatter output to [b*H+h][j][c] layout (QKV projections).
// ---------------------------------------------------------------------------
__global__ __launch_bounds__(256) void gemm_wmma_kernel(
    const float* __restrict__ A, const float* __restrict__ Bm,
    float* __restrict__ C, const float* __restrict__ bias,
    int M, int K, int Nc, int relu, int permute_head)
{
    __shared__ _Float16 lds_a[128 * 32];   // [m][k]
    __shared__ _Float16 lds_b[128 * 32];   // [n][k] (transposed)

    const int tid  = threadIdx.x;
    const int lane = tid & 31;
    const int wave = tid >> 5;
    const int wm   = wave >> 1;            // 0..3 -> 32 rows
    const int wn   = wave & 1;             // 0..1 -> 64 cols
    const int hf   = lane >> 4;
    const int ln   = lane & 15;
    const int m0   = blockIdx.x * 128;
    const int n0   = blockIdx.y * 128;

    v8f acc[2][4];
#pragma unroll
    for (int i = 0; i < 2; ++i)
#pragma unroll
        for (int j = 0; j < 4; ++j) acc[i][j] = v8f_zero();

    const int kfull = K >> 5;
    for (int ks = 0; ks < kfull; ++ks) {
        const int k0 = ks << 5;
        fx4 av[4], bv[4];
#pragma unroll
        for (int e = 0; e < 4; ++e) {          // A: 128x32 = 1024 fx4
            int idx = e * 256 + tid;
            int r = idx >> 3, kq = idx & 7;
            av[e] = *(const fx4*)&A[(size_t)(m0 + r) * K + k0 + kq * 4];
        }
#pragma unroll
        for (int e = 0; e < 4; ++e) {          // B: 32x128 = 1024 fx4
            int idx = e * 256 + tid;
            int kk = idx >> 5, nq = idx & 31;
            bv[e] = *(const fx4*)&Bm[(size_t)(k0 + kk) * Nc + n0 + nq * 4];
        }
#pragma unroll
        for (int e = 0; e < 4; ++e) {
            int idx = e * 256 + tid;
            int r = idx >> 3, kq = idx & 7;
            hx4 hv;
#pragma unroll
            for (int t = 0; t < 4; ++t) hv[t] = (_Float16)av[e][t];
            *(hx4*)&lds_a[r * 32 + kq * 4] = hv;
        }
#pragma unroll
        for (int e = 0; e < 4; ++e) {
            int idx = e * 256 + tid;
            int kk = idx >> 5, nq = idx & 31;
#pragma unroll
            for (int t = 0; t < 4; ++t)
                lds_b[(nq * 4 + t) * 32 + kk] = (_Float16)bv[e][t];
        }
        __syncthreads();

        v16h af[2], bf[4];
#pragma unroll
        for (int sm = 0; sm < 2; ++sm) {
            int m = wm * 32 + sm * 16 + ln;
            int base = hf * 8;
            h8 lo = *(const h8*)&lds_a[m * 32 + base];
            h8 hi = *(const h8*)&lds_a[m * 32 + base + 16];
            af[sm] = frag_cat(lo, hi);
        }
#pragma unroll
        for (int sn = 0; sn < 4; ++sn) {
            int n = wn * 64 + sn * 16 + ln;
            int koff = hf * 8;
            h8 lo = *(const h8*)&lds_b[n * 32 + koff];
            h8 hi = *(const h8*)&lds_b[n * 32 + koff + 16];
            bf[sn] = frag_cat(lo, hi);
        }
#pragma unroll
        for (int sm = 0; sm < 2; ++sm)
#pragma unroll
            for (int sn = 0; sn < 4; ++sn)
                acc[sm][sn] = wmma16(af[sm], bf[sn], acc[sm][sn]);
        __syncthreads();
    }

    if (K & 31) {                              // tail step (checked)
        const int k0 = kfull << 5;
        const int rem = K - k0;
#pragma unroll
        for (int e = 0; e < 16; ++e) {
            int idx = e * 256 + tid;
            int r = idx >> 5, kk = idx & 31;
            float v = (kk < rem) ? A[(size_t)(m0 + r) * K + k0 + kk] : 0.f;
            lds_a[r * 32 + kk] = (_Float16)v;
        }
#pragma unroll
        for (int e = 0; e < 16; ++e) {
            int idx = e * 256 + tid;
            int kk = idx >> 7, n = idx & 127;
            float v = (kk < rem) ? Bm[(size_t)(k0 + kk) * Nc + n0 + n] : 0.f;
            lds_b[n * 32 + kk] = (_Float16)v;
        }
        __syncthreads();
        v16h af[2], bf[4];
#pragma unroll
        for (int sm = 0; sm < 2; ++sm) {
            int m = wm * 32 + sm * 16 + ln;
            h8 lo = *(const h8*)&lds_a[m * 32 + hf * 8];
            h8 hi = *(const h8*)&lds_a[m * 32 + hf * 8 + 16];
            af[sm] = frag_cat(lo, hi);
        }
#pragma unroll
        for (int sn = 0; sn < 4; ++sn) {
            int n = wn * 64 + sn * 16 + ln;
            h8 lo = *(const h8*)&lds_b[n * 32 + hf * 8];
            h8 hi = *(const h8*)&lds_b[n * 32 + hf * 8 + 16];
            bf[sn] = frag_cat(lo, hi);
        }
#pragma unroll
        for (int sm = 0; sm < 2; ++sm)
#pragma unroll
            for (int sn = 0; sn < 4; ++sn)
                acc[sm][sn] = wmma16(af[sm], bf[sn], acc[sm][sn]);
    }

#pragma unroll
    for (int sm = 0; sm < 2; ++sm)
#pragma unroll
        for (int sn = 0; sn < 4; ++sn) {
            int n = n0 + wn * 64 + sn * 16 + ln;
            float bv = bias ? bias[n] : 0.f;
#pragma unroll
            for (int v = 0; v < 8; ++v) {
                int m = m0 + wm * 32 + sm * 16 + v + 8 * hf;
                float val = acc[sm][sn][v] + bv;
                if (relu) val = fmaxf(val, 0.f);
                if (permute_head) {
                    // [b][j][c*H+h] -> [(b*H+h)][j][c]
                    int bb = m >> 8, jr = m & 255, hh = n & 7, cc = n >> 3;
                    C[((size_t)(((bb << 3) + hh) << 8 | jr)) * CH_ + cc] = val;
                } else {
                    C[(size_t)m * Nc + n] = val;
                }
            }
        }
}

// ---------------------------------------------------------------------------
// Fused attention: one block per (b,h). 8 waves x 32 rows each.
// Q/K/V arrive in [(b*H+h)][j][c] layout (contiguous per head).
// ---------------------------------------------------------------------------
__global__ __launch_bounds__(256) void attn_kernel(
    const float* __restrict__ Qg, const float* __restrict__ Kg,
    const float* __restrict__ Vg, const float* __restrict__ Wb,
    const unsigned char* __restrict__ mask,
    float* __restrict__ a_out, float* __restrict__ cat)
{
    __shared__ _Float16 k_lds[N_ * CH_];      // [j][c]  32 KB
    __shared__ _Float16 vT_lds[CH_ * N_];     // [c][j]  32 KB
    __shared__ _Float16 a_lds[8][16 * N_];    // per-wave a rows, 64 KB
    __shared__ float    wb_lds[NB_];
    __shared__ unsigned char m_lds[N_];

    const int bh = blockIdx.x;
    const int b = bh / H_, h = bh % H_;
    const int tid  = threadIdx.x;
    const int lane = tid & 31;
    const int wave = tid >> 5;
    const int hf   = lane >> 4;
    const int ln   = lane & 15;

    const float* Qh = Qg + (size_t)bh * N_ * CH_;
    const float* Kh = Kg + (size_t)bh * N_ * CH_;
    const float* Vh = Vg + (size_t)bh * N_ * CH_;

#pragma unroll
    for (int e = 0; e < 16; ++e) {            // stage K (flat copy, f32->f16)
        int idx4 = e * 256 + tid;             // 4096 fx4
        fx4 v = ((const fx4*)Kh)[idx4];
        hx4 hv;
#pragma unroll
        for (int t = 0; t < 4; ++t) hv[t] = (_Float16)v[t];
        ((hx4*)k_lds)[idx4] = hv;
    }
#pragma unroll
    for (int e = 0; e < 16; ++e) {            // stage V^T
        int idx4 = e * 256 + tid;
        int j = idx4 >> 4, cq = idx4 & 15;    // Vh row j, cols cq*4..+3
        fx4 v = ((const fx4*)Vh)[idx4];
#pragma unroll
        for (int t = 0; t < 4; ++t) vT_lds[(cq * 4 + t) * N_ + j] = (_Float16)v[t];
    }
    if (tid < NB_) wb_lds[tid] = Wb[tid * H_ + h];
    m_lds[tid] = mask[b * N_ + tid];
    __syncthreads();

    const float invs = 0.125f;                 // 1/sqrt(CH)
    const float wl   = 0.70710678118654752f;   // sqrt(0.5)

    for (int rt = 0; rt < 2; ++rt) {
        const int i0 = (wave * 2 + rt) * 16;

        // ---- logits: q(16xCH) . k^T(CHxN) -> 16 x 256 strip ----
        v8f acc[16];
#pragma unroll
        for (int jt = 0; jt < 16; ++jt) acc[jt] = v8f_zero();

#pragma unroll
        for (int kk = 0; kk < 2; ++kk) {
            v16h af;
            {
                int m = i0 + ln;
                int base = kk * 32 + hf * 8;
                const float* qrow = &Qh[(size_t)m * CH_];
                fx4 q0 = *(const fx4*)&qrow[base];
                fx4 q1 = *(const fx4*)&qrow[base + 4];
                fx4 q2 = *(const fx4*)&qrow[base + 16];
                fx4 q3 = *(const fx4*)&qrow[base + 20];
#pragma unroll
                for (int t = 0; t < 4; ++t) {
                    af[t]      = (_Float16)q0[t];
                    af[t + 4]  = (_Float16)q1[t];
                    af[t + 8]  = (_Float16)q2[t];
                    af[t + 12] = (_Float16)q3[t];
                }
            }
#pragma unroll
            for (int jt = 0; jt < 16; ++jt) {
                int n = jt * 16 + ln;
                int koff = kk * 32 + hf * 8;
                h8 lo = *(const h8*)&k_lds[n * CH_ + koff];
                h8 hi = *(const h8*)&k_lds[n * CH_ + koff + 16];
                acc[jt] = wmma16(af, frag_cat(lo, hi), acc[jt]);
            }
        }

        // ---- bias + mask + softmax per row; a out; o_pair bins ----
#pragma unroll
        for (int v = 0; v < 8; ++v) {
            const int mrow = v + 8 * hf;
            const int i = i0 + mrow;
            const int mi = m_lds[i];
            float xv[16];
            float mx = -3.4e38f;
#pragma unroll
            for (int jt = 0; jt < 16; ++jt) {
                int j = jt * 16 + ln;
                int d = j - i;
                d = d < -KCLIP ? -KCLIP : (d > KCLIP ? KCLIP : d);
                float x;
                if (mi && m_lds[j]) x = wl * (acc[jt][v] * invs + wb_lds[d + KCLIP]);
                else                x = wl * (acc[jt][v] * invs) - 1e9f;
                xv[jt] = x;
                mx = fmaxf(mx, x);
            }
#pragma unroll
            for (int o = 1; o < 16; o <<= 1) mx = fmaxf(mx, __shfl_xor(mx, o));
            float sum = 0.f;
#pragma unroll
            for (int jt = 0; jt < 16; ++jt) { xv[jt] = __expf(xv[jt] - mx); sum += xv[jt]; }
#pragma unroll
            for (int o = 1; o < 16; o <<= 1) sum += __shfl_xor(sum, o);
            const float inv = 1.f / sum;

            float* arow = &a_out[(size_t)(bh * N_ + i) * N_];
            float p_[16];
#pragma unroll
            for (int jt = 0; jt < 16; ++jt) {
                int j = jt * 16 + ln;
                float p = xv[jt] * inv;
                p_[jt] = p;
                a_lds[wave][mrow * N_ + j] = (_Float16)p;
                arow[j] = p;
            }

            // o_pair: zero-fill, then interior single-writer + reduced edges.
            const size_t crow = (size_t)(b * N_ + i) * CATD + (size_t)h * NB_;
            for (int bb = ln; bb < NB_; bb += 16) cat[crow + bb] = 0.f;
            if (mi) {
                float s0 = 0.f, s1 = 0.f;
#pragma unroll
                for (int jt = 0; jt < 16; ++jt) {
                    int j = jt * 16 + ln;
                    int d = j - i;
                    if (d <= -KCLIP)      s0 += p_[jt];
                    else if (d >= KCLIP)  s1 += p_[jt];
                    else                  cat[crow + d + KCLIP] = p_[jt];
                }
#pragma unroll
                for (int o = 1; o < 16; o <<= 1) {
                    s0 += __shfl_xor(s0, o);
                    s1 += __shfl_xor(s1, o);
                }
                if (ln == 0) {
                    cat[crow] = s0;
                    cat[crow + NB_ - 1] = s1;
                }
            }
        }

        // ---- o = a @ v : 16 x CH ----
        v8f oacc[4];
#pragma unroll
        for (int ct = 0; ct < 4; ++ct) oacc[ct] = v8f_zero();
#pragma unroll
        for (int ks = 0; ks < 8; ++ks) {
            int base = ks * 32 + hf * 8;
            h8 alo = *(const h8*)&a_lds[wave][ln * N_ + base];
            h8 ahi = *(const h8*)&a_lds[wave][ln * N_ + base + 16];
            v16h af = frag_cat(alo, ahi);
#pragma unroll
            for (int ct = 0; ct < 4; ++ct) {
                int c = ct * 16 + ln;
                int koff = ks * 32 + hf * 8;
                h8 lo = *(const h8*)&vT_lds[c * N_ + koff];
                h8 hi = *(const h8*)&vT_lds[c * N_ + koff + 16];
                oacc[ct] = wmma16(af, frag_cat(lo, hi), oacc[ct]);
            }
        }
#pragma unroll
        for (int ct = 0; ct < 4; ++ct) {
            int c = ct * 16 + ln;
#pragma unroll
            for (int v = 0; v < 8; ++v) {
                int i = i0 + v + 8 * hf;
                cat[(size_t)(b * N_ + i) * CATD + NB_ * H_ + h * CH_ + c] = oacc[ct][v];
            }
        }
    }
}

// ---------------------------------------------------------------------------
// out = LayerNorm(X + Y) * g + b ; one 256-thread block per row of 256.
// ---------------------------------------------------------------------------
__global__ __launch_bounds__(256) void add_ln_kernel(
    const float* __restrict__ X, const float* __restrict__ Y,
    const float* __restrict__ g, const float* __restrict__ bta,
    float* __restrict__ out)
{
    __shared__ float red[8];
    const int row = blockIdx.x;
    const int tid = threadIdx.x;
    float x = X[(size_t)row * CS_ + tid] + Y[(size_t)row * CS_ + tid];

    float s = x;
#pragma unroll
    for (int o = 1; o < 32; o <<= 1) s += __shfl_xor(s, o);
    if ((tid & 31) == 0) red[tid >> 5] = s;
    __syncthreads();
    float tot = 0.f;
#pragma unroll
    for (int w = 0; w < 8; ++w) tot += red[w];
    const float mean = tot * (1.f / 256.f);
    const float d = x - mean;

    float s2 = d * d;
#pragma unroll
    for (int o = 1; o < 32; o <<= 1) s2 += __shfl_xor(s2, o);
    __syncthreads();
    if ((tid & 31) == 0) red[tid >> 5] = s2;
    __syncthreads();
    float var = 0.f;
#pragma unroll
    for (int w = 0; w < 8; ++w) var += red[w];
    var *= (1.f / 256.f);
    const float r = rsqrtf(var + 1e-5f);
    out[(size_t)row * CS_ + tid] = d * r * g[tid] + bta[tid];
}

// ---------------------------------------------------------------------------
extern "C" void kernel_launch(void* const* d_in, const int* in_sizes, int n_in,
                              void* d_out, int out_size, void* d_ws, size_t ws_size,
                              hipStream_t stream)
{
    (void)in_sizes; (void)n_in; (void)out_size; (void)ws_size;
    const float* s   = (const float*)d_in[0];
    const unsigned char* mask = (const unsigned char*)d_in[1];
    const float* Wq  = (const float*)d_in[2];
    const float* Wk  = (const float*)d_in[3];
    const float* Wv  = (const float*)d_in[4];
    const float* Wb  = (const float*)d_in[5];
    const float* Wo  = (const float*)d_in[6];
    const float* bo  = (const float*)d_in[7];
    const float* g1  = (const float*)d_in[8];
    const float* b1  = (const float*)d_in[9];
    const float* W1  = (const float*)d_in[10];
    const float* bf1 = (const float*)d_in[11];
    const float* W2  = (const float*)d_in[12];
    const float* bf2 = (const float*)d_in[13];
    const float* g2  = (const float*)d_in[14];
    const float* b2  = (const float*)d_in[15];

    float* out_s2 = (float*)d_out;
    float* out_a  = out_s2 + (size_t)B_ * N_ * CS_;

    const size_t M = (size_t)B_ * N_;   // 4096
    char* ws = (char*)d_ws;
    float* Qh   = (float*)ws; ws += M * QKN  * sizeof(float);
    float* Kh   = (float*)ws; ws += M * QKN  * sizeof(float);
    float* Vh   = (float*)ws; ws += M * QKN  * sizeof(float);
    float* cat  = (float*)ws; ws += M * CATD * sizeof(float);
    float* supd = (float*)ws; ws += M * CS_  * sizeof(float);
    float* s1   = (float*)ws; ws += M * CS_  * sizeof(float);
    float* ffh  = (float*)ws; ws += M * CT_  * sizeof(float);
    float* ff2  = (float*)ws; ws += M * CS_  * sizeof(float);

    dim3 blk(256);

    // Q, K, V projections -> head-major [(b*H+h)][j][c]
    gemm_wmma_kernel<<<dim3(32, QKN / 128), blk, 0, stream>>>(s, Wq, Qh, nullptr, 4096, CS_, QKN, 0, 1);
    gemm_wmma_kernel<<<dim3(32, QKN / 128), blk, 0, stream>>>(s, Wk, Kh, nullptr, 4096, CS_, QKN, 0, 1);
    gemm_wmma_kernel<<<dim3(32, QKN / 128), blk, 0, stream>>>(s, Wv, Vh, nullptr, 4096, CS_, QKN, 0, 1);

    // Attention (writes `a` output + cat = [o_pair | o])
    attn_kernel<<<dim3(B_ * H_), blk, 0, stream>>>(Qh, Kh, Vh, Wb, mask, out_a, cat);

    // s_upd = cat @ Wo + bo
    gemm_wmma_kernel<<<dim3(32, CS_ / 128), blk, 0, stream>>>(cat, Wo, supd, bo, 4096, CATD, CS_, 0, 0);
    // s1 = LN(s + s_upd)
    add_ln_kernel<<<dim3(4096), blk, 0, stream>>>(s, supd, g1, b1, s1);
    // ff hidden = relu(s1 @ W1 + bf1)
    gemm_wmma_kernel<<<dim3(32, CT_ / 128), blk, 0, stream>>>(s1, W1, ffh, bf1, 4096, CS_, CT_, 1, 0);
    // ff out = hidden @ W2 + bf2
    gemm_wmma_kernel<<<dim3(32, CS_ / 128), blk, 0, stream>>>(ffh, W2, ff2, bf2, 4096, CT_, CS_, 0, 0);
    // s2 = LN(s1 + ff)
    add_ln_kernel<<<dim3(4096), blk, 0, stream>>>(s1, ff2, g2, b2, out_s2);
}
